// ChemicalConstraintGNN_33887291965650
// MI455X (gfx1250) — compile-verified, matrix operand order checked by reading, not run on hardware
//
#include <hip/hip_runtime.h>
#include <hip/hip_bf16.h>

// ---------------------------------------------------------------------------
// ChemicalConstraintGNN on MI455X (gfx1250)
//   - message MLP (E x [514->256->256]) and update MLP (N x [512->256->256])
//     through v_wmma_f32_16x16x32_bf16 (fp32 accum, bf16 operands)
//   - B fragments reused across 8 row tiles + double-buffered across K-chunks
//   - A staging: coalesced b128 gathers -> registers -> ping-pong LDS (bf16)
//   - LDS strides padded for bank-conflict-free fragment reads
// ---------------------------------------------------------------------------

#define ND    256      // D
#define NN    10000    // nodes
#define NE    320000   // edges
#define KM1   514      // 2D+2
#define KM1C  17       // ceil(514/32)
#define KU1C  16       // 512/32
#define KDC   8        // 256/32
#define LMS   40       // staged-chunk LDS row stride (elems, pad 32->40)
#define LHS   264      // hidden-tile LDS row stride (elems, pad 256->264)

typedef __attribute__((ext_vector_type(16))) __bf16 bf16x16;
typedef __attribute__((ext_vector_type(8)))  __bf16 bf16x8;
typedef __attribute__((ext_vector_type(8)))  float  f32x8;

union BfFrag { bf16x16 v; bf16x8 h2[2]; };

__device__ __forceinline__ f32x8 wmma_bf16(bf16x16 a, bf16x16 b, f32x8 c) {
  // (neg_a, A, neg_b, B, c_mod, C, reuse_a, reuse_b)
  return __builtin_amdgcn_wmma_f32_16x16x32_bf16(false, a, false, b, (short)0, c,
                                                 false, false);
}

__device__ __forceinline__ BfFrag load_b_frag(const __bf16* __restrict__ Ws,
                                              int kc, int nt, int lane) {
  const __bf16* bp = Ws + (((size_t)(kc * 16 + nt)) * 32 + lane) * 16;
  BfFrag b;
  b.h2[0] = *(const bf16x8*)bp;
  b.h2[1] = *(const bf16x8*)(bp + 8);
  return b;
}

__device__ __forceinline__ BfFrag load_a_frag(const __bf16* ap) {
  BfFrag a;
  a.h2[0] = *(const bf16x8*)ap;
  a.h2[1] = *(const bf16x8*)(ap + 16);
  return a;
}

// ---------------------------------------------------------------------------
__global__ void zero_kernel(float* __restrict__ p, int n) {
  int i = blockIdx.x * blockDim.x + threadIdx.x;
  if (i < n) p[i] = 0.f;
}

// Row-major fp32 W[Ktot x 256] -> swizzled bf16 B-fragments.
// dst[((kc*16 + nt)*32 + lane)*16 + i]; n = nt*16+(lane&15);
// k = kc*32 + (lane>=16?8:0) + (i<8 ? i : 8+i)
__global__ void swizzle_weights_kernel(const float* __restrict__ W,
                                       __bf16* __restrict__ Ws,
                                       int Ktot, int total) {
  int t = blockIdx.x * blockDim.x + threadIdx.x;
  if (t >= total) return;
  int i    = t & 15;
  int lane = (t >> 4) & 31;
  int nt   = (t >> 9) & 15;
  int kc   = t >> 13;
  int b    = (lane >= 16) ? 8 : 0;
  int k    = kc * 32 + b + ((i < 8) ? i : (8 + i));
  int n    = nt * 16 + (lane & 15);
  float v  = (k < Ktot) ? W[(size_t)k * ND + n] : 0.f;
  Ws[t] = (__bf16)v;
}

// valence attention: sigmoid(relu([vr,vc]@Wa1+ba1)@Wa2+ba2) per edge
__global__ void attention_kernel(const int* __restrict__ ei,
                                 const int* __restrict__ valences,
                                 const float* __restrict__ Wa1,
                                 const float* __restrict__ ba1,
                                 const float* __restrict__ Wa2,
                                 const float* __restrict__ ba2,
                                 float* __restrict__ att) {
  int e = blockIdx.x * blockDim.x + threadIdx.x;
  if (e >= NE) return;
  float vr = (float)valences[ei[e]];
  float vc = (float)valences[ei[NE + e]];
  float s = ba2[0];
#pragma unroll
  for (int j = 0; j < 16; ++j) {
    float a = vr * Wa1[j] + vc * Wa1[16 + j] + ba1[j];
    a = a > 0.f ? a : 0.f;
    s += a * Wa2[j];
  }
  att[e] = 1.f / (1.f + __expf(-s));
}

// ---------------------------------------------------------------------------
// Fused per-edge message MLP + attention scale + atomic scatter-add.
// 128 edges / WG. 8 waves; wave w owns N-tiles {2w, 2w+1} for ALL 8 row-tiles.
__global__ void __launch_bounds__(256)
edge_message_kernel(const float* __restrict__ h, const int* __restrict__ ei,
                    const int* __restrict__ valences,
                    const __bf16* __restrict__ Wm1s, const float* __restrict__ bm1,
                    const __bf16* __restrict__ Wm2s, const float* __restrict__ bm2,
                    const float* __restrict__ att, float* __restrict__ h_agg) {
  constexpr int RT = 8, TMR = RT * 16;      // 128 edges per workgroup
  __shared__ __bf16 lmsg[2][TMR * LMS];     // ping-pong staged K-chunk
  __shared__ __bf16 lhid[TMR * LHS];        // layer-1 activations (bf16)
  __shared__ int    lrow[TMR], lcol[TMR];
  __shared__ float  lvr[TMR], lvc[TMR], latt[TMR];

  const int tid  = threadIdx.x;
  const int base = blockIdx.x * TMR;

  if (tid < TMR) {
    int e = base + tid;
    if (e < NE) {
      int r = ei[e], c = ei[NE + e];
      lrow[tid] = r; lcol[tid] = c;
      lvr[tid]  = (float)valences[r];
      lvc[tid]  = (float)valences[c];
      latt[tid] = att[e];
    } else {
      lrow[tid] = -1; lcol[tid] = 0;
      lvr[tid] = 0.f; lvc[tid] = 0.f; latt[tid] = 0.f;
    }
  }
  __syncthreads();

  const int lane = tid & 31, wave = tid >> 5;
  const int lx   = lane & 15;
  const int hb   = (lane >> 4) << 3;        // 0 or 8 (K sub-stripe)
  const int nt0  = wave * 2;

  // staging: thread owns 16 contiguous cols of row (tid>>1)
  const int  srow = tid >> 1, shalf = tid & 1;
  const int  sr = lrow[srow], sc = lcol[srow];
  const float svr = lvr[srow], svc = lvc[srow];

  float pre[16];
  auto gather = [&](int kc) {
    int gcb = kc * 32 + shalf * 16;
    const float* src = nullptr;
    if (sr >= 0) {
      if (gcb < ND)          src = h + (size_t)sr * ND + gcb;
      else if (gcb < 2 * ND) src = h + (size_t)sc * ND + (gcb - ND);
    }
    if (src) {
      const float4* p4 = (const float4*)src;
      float4 q0 = p4[0], q1 = p4[1], q2 = p4[2], q3 = p4[3];
      pre[0]=q0.x;  pre[1]=q0.y;  pre[2]=q0.z;  pre[3]=q0.w;
      pre[4]=q1.x;  pre[5]=q1.y;  pre[6]=q1.z;  pre[7]=q1.w;
      pre[8]=q2.x;  pre[9]=q2.y;  pre[10]=q2.z; pre[11]=q2.w;
      pre[12]=q3.x; pre[13]=q3.y; pre[14]=q3.z; pre[15]=q3.w;
    } else {
#pragma unroll
      for (int j = 0; j < 16; ++j) {
        int gc = gcb + j;
        float v = 0.f;
        if (sr >= 0) { if (gc == 2*ND) v = svr; else if (gc == 2*ND+1) v = svc; }
        pre[j] = v;
      }
    }
  };
  auto commit = [&](int buf) {
    bf16x8 p0, p1;
#pragma unroll
    for (int j = 0; j < 8; ++j) { p0[j] = (__bf16)pre[j]; p1[j] = (__bf16)pre[8+j]; }
    __bf16* dst = &lmsg[buf][srow * LMS + shalf * 16];
    *(bf16x8*)dst       = p0;
    *(bf16x8*)(dst + 8) = p1;
  };

  // ---------------- GEMM1: msg_in[128 x 514] @ Wm1 ----------------
  gather(0);
  BfFrag b0 = load_b_frag(Wm1s, 0, nt0,     lane);
  BfFrag b1 = load_b_frag(Wm1s, 0, nt0 + 1, lane);

  f32x8 acc[RT][2] = {};

  for (int kc = 0; kc < KM1C; ++kc) {
    int buf = kc & 1;
    commit(buf);
    __syncthreads();
    BfFrag nb0 = b0, nb1 = b1;
    if (kc + 1 < KM1C) {
      gather(kc + 1);                              // in-flight across compute
      nb0 = load_b_frag(Wm1s, kc + 1, nt0,     lane);
      nb1 = load_b_frag(Wm1s, kc + 1, nt0 + 1, lane);
    }
    const __bf16* abase = &lmsg[buf][lx * LMS + hb];
#pragma unroll
    for (int rt = 0; rt < RT; ++rt) {
      BfFrag a = load_a_frag(abase + rt * 16 * LMS);
      acc[rt][0] = wmma_bf16(a.v, b0.v, acc[rt][0]);
      acc[rt][1] = wmma_bf16(a.v, b1.v, acc[rt][1]);
    }
    b0 = nb0; b1 = nb1;
  }

  // bias + ReLU -> bf16 hidden tile
#pragma unroll
  for (int nt = 0; nt < 2; ++nt) {
    int n = (nt0 + nt) * 16 + lx;
    float bia = bm1[n];
#pragma unroll
    for (int rt = 0; rt < RT; ++rt)
#pragma unroll
      for (int v = 0; v < 8; ++v) {
        float x = acc[rt][nt][v] + bia;
        lhid[(rt * 16 + hb + v) * LHS + n] = (__bf16)(x > 0.f ? x : 0.f);
      }
  }
  __syncthreads();

  // ---------------- GEMM2: hid[128 x 256] @ Wm2 ----------------
  f32x8 acc2[RT][2] = {};
  BfFrag c0 = load_b_frag(Wm2s, 0, nt0,     lane);
  BfFrag c1 = load_b_frag(Wm2s, 0, nt0 + 1, lane);
  for (int kc = 0; kc < KDC; ++kc) {
    BfFrag n0 = c0, n1 = c1;
    if (kc + 1 < KDC) {
      n0 = load_b_frag(Wm2s, kc + 1, nt0,     lane);
      n1 = load_b_frag(Wm2s, kc + 1, nt0 + 1, lane);
    }
    const __bf16* abase = &lhid[lx * LHS + kc * 32 + hb];
#pragma unroll
    for (int rt = 0; rt < RT; ++rt) {
      BfFrag a = load_a_frag(abase + rt * 16 * LHS);
      acc2[rt][0] = wmma_bf16(a.v, c0.v, acc2[rt][0]);
      acc2[rt][1] = wmma_bf16(a.v, c1.v, acc2[rt][1]);
    }
    c0 = n0; c1 = n1;
  }

  // (acc + bm2) * att -> atomic scatter-add into h_agg[row]
#pragma unroll
  for (int nt = 0; nt < 2; ++nt) {
    int n = (nt0 + nt) * 16 + lx;
    float bia = bm2[n];
#pragma unroll
    for (int rt = 0; rt < RT; ++rt)
#pragma unroll
      for (int v = 0; v < 8; ++v) {
        int eloc = rt * 16 + hb + v;
        int r = lrow[eloc];
        if (r >= 0) {
          float x = (acc2[rt][nt][v] + bia) * latt[eloc];
          atomicAdd(&h_agg[(size_t)r * ND + n], x);
        }
      }
  }
}

// ---------------------------------------------------------------------------
// Node update MLP on [h | h_agg] + residual.  64 nodes / WG, RT=4.
__global__ void __launch_bounds__(256)
node_update_kernel(const float* __restrict__ h, const float* __restrict__ h_agg,
                   const __bf16* __restrict__ Wu1s, const float* __restrict__ bu1,
                   const __bf16* __restrict__ Wu2s, const float* __restrict__ bu2,
                   float* __restrict__ out) {
  constexpr int RT = 4, TMR = RT * 16;      // 64 nodes per workgroup
  __shared__ __bf16 lin[2][TMR * LMS];
  __shared__ __bf16 lhid[TMR * LHS];

  const int tid  = threadIdx.x;
  const int base = blockIdx.x * TMR;
  const int lane = tid & 31, wave = tid >> 5;
  const int lx   = lane & 15;
  const int hb   = (lane >> 4) << 3;
  const int nt0  = wave * 2;

  // staging: thread owns 8 contiguous cols of row (tid>>2)
  const int srow = tid >> 2, sq = tid & 3;
  const int snode = base + srow;

  float pre[8];
  auto gather = [&](int kc) {
    int gcb = kc * 32 + sq * 8;
    if (snode < NN) {
      const float* src = (gcb < ND) ? h + (size_t)snode * ND + gcb
                                    : h_agg + (size_t)snode * ND + (gcb - ND);
      const float4* p4 = (const float4*)src;
      float4 q0 = p4[0], q1 = p4[1];
      pre[0]=q0.x; pre[1]=q0.y; pre[2]=q0.z; pre[3]=q0.w;
      pre[4]=q1.x; pre[5]=q1.y; pre[6]=q1.z; pre[7]=q1.w;
    } else {
#pragma unroll
      for (int j = 0; j < 8; ++j) pre[j] = 0.f;
    }
  };
  auto commit = [&](int buf) {
    bf16x8 p0;
#pragma unroll
    for (int j = 0; j < 8; ++j) p0[j] = (__bf16)pre[j];
    *(bf16x8*)&lin[buf][srow * LMS + sq * 8] = p0;
  };

  gather(0);
  BfFrag b0 = load_b_frag(Wu1s, 0, nt0,     lane);
  BfFrag b1 = load_b_frag(Wu1s, 0, nt0 + 1, lane);

  f32x8 acc[RT][2] = {};
  for (int kc = 0; kc < KU1C; ++kc) {
    int buf = kc & 1;
    commit(buf);
    __syncthreads();
    BfFrag nb0 = b0, nb1 = b1;
    if (kc + 1 < KU1C) {
      gather(kc + 1);
      nb0 = load_b_frag(Wu1s, kc + 1, nt0,     lane);
      nb1 = load_b_frag(Wu1s, kc + 1, nt0 + 1, lane);
    }
    const __bf16* abase = &lin[buf][lx * LMS + hb];
#pragma unroll
    for (int rt = 0; rt < RT; ++rt) {
      BfFrag a = load_a_frag(abase + rt * 16 * LMS);
      acc[rt][0] = wmma_bf16(a.v, b0.v, acc[rt][0]);
      acc[rt][1] = wmma_bf16(a.v, b1.v, acc[rt][1]);
    }
    b0 = nb0; b1 = nb1;
  }

#pragma unroll
  for (int nt = 0; nt < 2; ++nt) {
    int n = (nt0 + nt) * 16 + lx;
    float bia = bu1[n];
#pragma unroll
    for (int rt = 0; rt < RT; ++rt)
#pragma unroll
      for (int v = 0; v < 8; ++v) {
        float x = acc[rt][nt][v] + bia;
        lhid[(rt * 16 + hb + v) * LHS + n] = (__bf16)(x > 0.f ? x : 0.f);
      }
  }
  __syncthreads();

  f32x8 acc2[RT][2] = {};
  BfFrag c0 = load_b_frag(Wu2s, 0, nt0,     lane);
  BfFrag c1 = load_b_frag(Wu2s, 0, nt0 + 1, lane);
  for (int kc = 0; kc < KDC; ++kc) {
    BfFrag n0 = c0, n1 = c1;
    if (kc + 1 < KDC) {
      n0 = load_b_frag(Wu2s, kc + 1, nt0,     lane);
      n1 = load_b_frag(Wu2s, kc + 1, nt0 + 1, lane);
    }
    const __bf16* abase = &lhid[lx * LHS + kc * 32 + hb];
#pragma unroll
    for (int rt = 0; rt < RT; ++rt) {
      BfFrag a = load_a_frag(abase + rt * 16 * LHS);
      acc2[rt][0] = wmma_bf16(a.v, c0.v, acc2[rt][0]);
      acc2[rt][1] = wmma_bf16(a.v, c1.v, acc2[rt][1]);
    }
    c0 = n0; c1 = n1;
  }

#pragma unroll
  for (int nt = 0; nt < 2; ++nt) {
    int n = (nt0 + nt) * 16 + lx;
    float bia = bu2[n];
#pragma unroll
    for (int rt = 0; rt < RT; ++rt)
#pragma unroll
      for (int v = 0; v < 8; ++v) {
        int node = base + rt * 16 + hb + v;
        if (node < NN)
          out[(size_t)node * ND + n] = h[(size_t)node * ND + n] + acc2[rt][nt][v] + bia;
      }
  }
}

// ---------------------------------------------------------------------------
extern "C" void kernel_launch(void* const* d_in, const int* in_sizes, int n_in,
                              void* d_out, int out_size, void* d_ws, size_t ws_size,
                              hipStream_t stream) {
  (void)in_sizes; (void)n_in; (void)out_size; (void)ws_size;
  const float* h        = (const float*)d_in[0];
  const int*   ei       = (const int*)d_in[1];
  /* d_in[2] edge_attr: unused by reference */
  const int*   valences = (const int*)d_in[3];
  const float* Wm1 = (const float*)d_in[4];
  const float* bm1 = (const float*)d_in[5];
  const float* Wm2 = (const float*)d_in[6];
  const float* bm2 = (const float*)d_in[7];
  const float* Wu1 = (const float*)d_in[8];
  const float* bu1 = (const float*)d_in[9];
  const float* Wu2 = (const float*)d_in[10];
  const float* bu2 = (const float*)d_in[11];
  const float* Wa1 = (const float*)d_in[12];
  const float* ba1 = (const float*)d_in[13];
  const float* Wa2 = (const float*)d_in[14];
  const float* ba2 = (const float*)d_in[15];
  float* out = (float*)d_out;

  char* ws = (char*)d_ws;
  size_t off = 0;
  float*  att   = (float*)(ws + off);  off += (size_t)NE * 4;            // 1.28 MB
  float*  h_agg = (float*)(ws + off);  off += (size_t)NN * ND * 4;       // 10.24 MB
  __bf16* Wm1s  = (__bf16*)(ws + off); off += (size_t)KM1C * 8192 * 2;   // 272 KB
  __bf16* Wm2s  = (__bf16*)(ws + off); off += (size_t)KDC  * 8192 * 2;   // 128 KB
  __bf16* Wu1s  = (__bf16*)(ws + off); off += (size_t)KU1C * 8192 * 2;   // 256 KB
  __bf16* Wu2s  = (__bf16*)(ws + off); off += (size_t)KDC  * 8192 * 2;   // 128 KB

  zero_kernel<<<(NN * ND + 255) / 256, 256, 0, stream>>>(h_agg, NN * ND);

  swizzle_weights_kernel<<<(KM1C * 8192 + 255) / 256, 256, 0, stream>>>(Wm1, Wm1s, KM1, KM1C * 8192);
  swizzle_weights_kernel<<<(KDC  * 8192 + 255) / 256, 256, 0, stream>>>(Wm2, Wm2s, ND,  KDC  * 8192);
  swizzle_weights_kernel<<<(KU1C * 8192 + 255) / 256, 256, 0, stream>>>(Wu1, Wu1s, 2 * ND, KU1C * 8192);
  swizzle_weights_kernel<<<(KDC  * 8192 + 255) / 256, 256, 0, stream>>>(Wu2, Wu2s, ND,  KDC  * 8192);

  attention_kernel<<<(NE + 255) / 256, 256, 0, stream>>>(ei, valences, Wa1, ba1, Wa2, ba2, att);

  edge_message_kernel<<<(NE + 127) / 128, 256, 0, stream>>>(
      h, ei, valences, Wm1s, bm1, Wm2s, bm2, att, h_agg);

  node_update_kernel<<<(NN + 63) / 64, 256, 0, stream>>>(
      h, h_agg, Wu1s, bu1, Wu2s, bu2, out);
}
